// VectorQuantizer_14465449853132
// MI455X (gfx1250) — matrix operand order repaired and checked by reading, not compiled, not run on hardware
//
#include <hip/hip_runtime.h>
#include <hip/hip_bf16.h>

// ---------------------------------------------------------------------------
// VQ-VAE codebook update for MI455X (gfx1250, wave32, WMMA bf16).
//   argmin ||x-e||^2  ==  argmax( x.e - 0.5||e||^2 )
// Distance GEMM runs on v_wmma_f32_16x16x32_bf16 with register-resident A
// (x rows) and double-buffered LDS B tiles streamed by the CDNA5 async
// global->LDS DMA path (global_load_async_to_lds_b128 / ASYNCcnt).
// Codebook is pre-converted to bf16 and pre-swizzled into B-fragment layout
// once, so the hot loop does zero conversion VALU work.
// ---------------------------------------------------------------------------

typedef __attribute__((ext_vector_type(16))) __bf16 v16bf;
typedef __attribute__((ext_vector_type(8)))  float  v8f;

#define BATCH 16
#define DIM   512
#define TLEN  4096
#define KCB   1024
#define NVEC  (BATCH * TLEN)          // 65536
#define DECAY 0.99f
#define EPSV  1e-5f

__device__ __forceinline__ unsigned short bfbits(float f) {
    return __builtin_bit_cast(unsigned short, (__bf16)f);
}

__device__ __forceinline__ void wait_async0() {
#if __has_builtin(__builtin_amdgcn_s_wait_asynccnt)
    __builtin_amdgcn_s_wait_asynccnt(0);
#else
    asm volatile("s_wait_asynccnt 0x0" ::: "memory");
#endif
}

// ---------------------------------------------------------------------------
// 0.5 * ||codebook[k]||^2
__global__ __launch_bounds__(128) void vq_cnorm(const float* __restrict__ cb,
                                                float* __restrict__ cnorm) {
    __shared__ float red[128];
    const int k = blockIdx.x, tid = threadIdx.x;
    const float* row = cb + (size_t)k * DIM;
    float s = 0.f;
    for (int d = tid; d < DIM; d += 128) { float v = row[d]; s += v * v; }
    red[tid] = s; __syncthreads();
    for (int off = 64; off > 0; off >>= 1) {
        if (tid < off) red[tid] += red[tid + off];
        __syncthreads();
    }
    if (tid == 0) cnorm[k] = 0.5f * red[0];
}

__global__ void vq_zero(float* __restrict__ p, int n) {
    int i = blockIdx.x * 256 + threadIdx.x;
    if (i < n) p[i] = 0.f;
}

// ---------------------------------------------------------------------------
// One-shot: codebook fp32 [1024,512] -> bf16 B-fragment stream
// cbfrag[ct][fr][i], fr = kk*32 + lane_slot, i = 0..15 packed as u32 pairs.
// Each 16-code column tile becomes a contiguous 16 KB block => async DMA-able.
__global__ __launch_bounds__(256) void vq_bswz(const float* __restrict__ cb,
                                               unsigned* __restrict__ cbfrag) {
    const int e  = blockIdx.x * 256 + threadIdx.x;   // < K*D/2 = 262144
    const int p  = e & 7;                            // bf16 pair within frag row
    const int fr = (e >> 3) & 511;                   // kk*32 + lane_slot
    const int ct = e >> 12;
    const int kk = fr >> 5, ls = fr & 31;
    const int code = ct * 16 + (ls & 15);
    const int d    = kk * 32 + (ls >> 4) * 8 + (p >> 2) * 8 + 2 * p;
    const float* src = cb + (size_t)code * DIM + d;
    cbfrag[e] = (unsigned)bfbits(src[0]) | ((unsigned)bfbits(src[1]) << 16);
}

// ---------------------------------------------------------------------------
// Nearest-code search. One wave = 16 x-rows; A fragments (16x512 bf16) are
// register resident in the ISA 16-bit A 16x32 layout. B tiles (16 KB each)
// are double-buffered in LDS via async global->LDS loads.
__global__ __launch_bounds__(256) void vq_argmin(const float* __restrict__ x,
                                                 const unsigned short* __restrict__ cbfrag,
                                                 const float* __restrict__ cnorm,
                                                 int* __restrict__ idx_out) {
    __shared__ __align__(128) unsigned short bsh[2][16 * 32 * 16];  // 2 x 16 KB

    const int tid  = threadIdx.x;
    const int lane = tid & 31;
    const int wave = tid >> 5;
    const int r    = lane & 15;     // A row within tile / C column
    const int kh   = lane >> 4;     // lane half
    const int kb   = kh * 8;        // A k-base per ISA layout

    const int rowbase = (blockIdx.x * 8 + wave) * 16;   // global x-row tile
    const int bb = rowbase >> 12;                       // /4096 -> batch
    const int t0 = rowbase & 4095;
    const float* xb = x + (size_t)bb * DIM * TLEN + (t0 + r);

    // async-stage one 16 KB B tile: 4 x b128 per thread (block covers tile)
    const unsigned long long gbase =
        (unsigned long long)(uintptr_t)cbfrag + (unsigned)tid * 16u;
    const unsigned lds0 = (unsigned)(uintptr_t)&bsh[0][0] + (unsigned)tid * 16u;
    auto stage = [&](int ct, int buf) {
        unsigned long long g = gbase + (unsigned long long)ct * 16384ull;
        unsigned l = lds0 + (unsigned)buf * 16384u;
#pragma unroll
        for (int j = 0; j < 4; ++j) {
            asm volatile("global_load_async_to_lds_b128 %0, %1, off"
                         :: "v"(l + (unsigned)j * 4096u),
                            "v"(g + (unsigned long long)j * 4096ull)
                         : "memory");
        }
    };

    // ---- load A fragments (x rows) into registers, fp32 -> bf16 ----
    stage(0, 0);
    v16bf af[16];
#pragma unroll
    for (int kk = 0; kk < 16; ++kk) {
#pragma unroll
        for (int i = 0; i < 16; ++i) {
            const int d = 32 * kk + (i < 8 ? (kb + i) : (16 + kb + (i - 8)));
            af[kk][i] = (__bf16)xb[(size_t)d * TLEN];
        }
    }

    float best[8];
    int   bidx[8];
#pragma unroll
    for (int v = 0; v < 8; ++v) { best[v] = -3.4e38f; bidx[v] = 0; }

    const int c = lane & 15;        // C-matrix column owned by this lane

    for (int ct = 0; ct < 64; ++ct) {
        const int buf = ct & 1;
        wait_async0();              // our wave's stage of tile ct is done
        __syncthreads();            // => every wave's portion is done
        if (ct + 1 < 64) stage(ct + 1, buf ^ 1);   // prefetch next tile

        // ---- 16x16 score tile: acc = x . e over D=512 (16 WMMAs) ----
        v8f acc = {};
#pragma unroll
        for (int kk = 0; kk < 16; ++kk) {
            v16bf bfrag = *(const v16bf*)&bsh[buf][(kk * 32 + lane) * 16];
            acc = __builtin_amdgcn_wmma_f32_16x16x32_bf16(
                false, af[kk], false, bfrag, (short)0, acc, false, false);
        }

        // ---- running argmax of (x.e - 0.5||e||^2) ----
        const int code = ct * 16 + c;
        const float cn = cnorm[code];
#pragma unroll
        for (int v = 0; v < 8; ++v) {
            const float s = acc[v] - cn;
            if (s > best[v] || (s == best[v] && code < bidx[v])) {
                best[v] = s; bidx[v] = code;
            }
        }
    }

    // ---- reduce across the 16 lanes holding each row's candidates ----
#pragma unroll
    for (int v = 0; v < 8; ++v) {
#pragma unroll
        for (int m = 1; m < 16; m <<= 1) {
            const float ov = __shfl_xor(best[v], m, 32);
            const int   oi = __shfl_xor(bidx[v], m, 32);
            if (ov > best[v] || (ov == best[v] && oi < bidx[v])) {
                best[v] = ov; bidx[v] = oi;
            }
        }
        if (c == 0) idx_out[rowbase + v + 8 * kh] = bidx[v];
    }
}

// ---------------------------------------------------------------------------
// segment sums: cluster_size[k] += 1, cluster_sum[k][d] += x.  Coalesced
// (b,d)-slab reads of x, scattered f32 atomics into L2.
__global__ __launch_bounds__(256) void vq_scatter(const float* __restrict__ x,
                                                  const int* __restrict__ idx,
                                                  float* __restrict__ csum,
                                                  float* __restrict__ csize) {
    const int bd = blockIdx.x;          // b*512 + d
    const int bb = bd >> 9;
    const int d  = bd & 511;
    const float* xr = x + (size_t)bd * TLEN;
    const int*   ir = idx + bb * TLEN;
    for (int j = 0; j < 16; ++j) {
        const int t = threadIdx.x + j * 256;
        const int k = ir[t];
        atomicAdd(&csum[(size_t)k * DIM + d], xr[t]);
        if (d == 0) atomicAdd(&csize[k], 1.0f);
    }
}

// ---------------------------------------------------------------------------
// EMA size update + n-sum + perplexity (single 1024-thread block).
__global__ __launch_bounds__(1024) void vq_stats(const float* __restrict__ ema_size,
                                                 const float* __restrict__ csize,
                                                 float* __restrict__ nes_out,
                                                 float* __restrict__ scal,
                                                 float* __restrict__ perp_out) {
    __shared__ float red[1024];
    const int k = threadIdx.x;
    const float cs  = csize[k];
    const float nes = ema_size[k] * DECAY + (1.0f - DECAY) * cs;
    nes_out[k] = nes;

    red[k] = nes; __syncthreads();
    for (int off = 512; off > 0; off >>= 1) {
        if (k < off) red[k] += red[k + off];
        __syncthreads();
    }
    const float nsum = red[0];
    __syncthreads();

    red[k] = cs; __syncthreads();
    for (int off = 512; off > 0; off >>= 1) {
        if (k < off) red[k] += red[k + off];
        __syncthreads();
    }
    const float tot = red[0];
    __syncthreads();

    const float p = cs / fmaxf(tot, 1.0f);
    red[k] = p * logf(p + 1e-10f);
    __syncthreads();
    for (int off = 512; off > 0; off >>= 1) {
        if (k < off) red[k] += red[k + off];
        __syncthreads();
    }
    if (k == 0) {
        scal[0] = nsum;
        perp_out[0] = expf(-red[0]);
    }
}

// ---------------------------------------------------------------------------
// new_ema_sum + smoothed normalization -> new_codebook
__global__ __launch_bounds__(256) void vq_codebook(const float* __restrict__ ema_sum,
                                                   const float* __restrict__ csum,
                                                   const float* __restrict__ nes,
                                                   const float* __restrict__ scal,
                                                   float* __restrict__ ncb_out,
                                                   float* __restrict__ nsum_out) {
    const int i = blockIdx.x * 256 + threadIdx.x;   // < K*D
    const int k = i >> 9;
    const float ncs = ema_sum[i] * DECAY + (1.0f - DECAY) * csum[i];
    nsum_out[i] = ncs;
    const float n  = scal[0];
    const float sm = (nes[k] + EPSV) / (n + (float)KCB * EPSV) * n;
    ncb_out[i] = ncs / sm;
}

// ---------------------------------------------------------------------------
// quantized = new_codebook[idx] (written back in [b,d,t] layout) + commit loss
__global__ __launch_bounds__(256) void vq_gather(const float* __restrict__ x,
                                                 const float* __restrict__ ncb,
                                                 const int* __restrict__ idx,
                                                 float* __restrict__ qout,
                                                 float* __restrict__ loss_acc) {
    __shared__ float red[256];
    const int bd = blockIdx.x;          // b*512 + d
    const int bb = bd >> 9;
    const int d  = bd & 511;
    const float* xr = x + (size_t)bd * TLEN;
    float*       qr = qout + (size_t)bd * TLEN;
    const int*   ir = idx + bb * TLEN;
    float ls = 0.f;
    for (int j = 0; j < 16; ++j) {
        const int t = threadIdx.x + j * 256;
        const int k = ir[t];
        const float q  = ncb[(size_t)k * DIM + d];
        const float xv = xr[t];
        qr[t] = q;
        const float df = xv - q;
        ls += df * df;
    }
    red[threadIdx.x] = ls; __syncthreads();
    for (int off = 128; off > 0; off >>= 1) {
        if (threadIdx.x < off) red[threadIdx.x] += red[threadIdx.x + off];
        __syncthreads();
    }
    if (threadIdx.x == 0) atomicAdd(loss_acc, red[0]);
}

// ---------------------------------------------------------------------------
__global__ void vq_final(const int* __restrict__ idx, const float* __restrict__ scal,
                         float* __restrict__ ind_out, float* __restrict__ loss_out) {
    const int i = blockIdx.x * 256 + threadIdx.x;
    if (i < NVEC) ind_out[i] = (float)idx[i];
    if (i == 0) loss_out[0] = scal[1] * (1.0f / ((float)BATCH * DIM * TLEN));
}

// ---------------------------------------------------------------------------
extern "C" void kernel_launch(void* const* d_in, const int* in_sizes, int n_in,
                              void* d_out, int out_size, void* d_ws, size_t ws_size,
                              hipStream_t stream) {
    const float* x        = (const float*)d_in[0];
    const float* cbw      = (const float*)d_in[1];
    const float* ema_size = (const float*)d_in[2];
    const float* ema_sum  = (const float*)d_in[3];

    float* out = (float*)d_out;
    float* q_out    = out;                      // [16,512,4096]
    float* ind_out  = out + 33554432;           // [16,4096] (indices as float)
    float* loss_out = out + 33619968;           // scalar
    float* perp_out = out + 33619969;           // scalar
    float* ncb_out  = out + 33619970;           // [1024,512]
    float* nes_out  = out + 34144258;           // [1024]
    float* nsum_out = out + 34145282;           // [1024,512]

    float*    ws     = (float*)d_ws;
    float*    cnormp = ws;                      // 1024
    float*    csizep = ws + 1024;               // 1024
    float*    csump  = ws + 2048;               // 524288
    float*    scalp  = ws + 526336;             // [0]=n sum, [1]=loss sum
    int*      idxp   = (int*)(ws + 526344);     // 65536 ints
    unsigned* cbfr   = (unsigned*)(ws + 591880);// 262144 u32 = 1 MB bf16 frags

    // zero csize + csum + scalars (idx/cnorm/cbfrag fully overwritten each call)
    const int nz = 1024 + KCB * DIM + 8;
    vq_zero    <<<(nz + 255) / 256, 256, 0, stream>>>(ws + 1024, nz);
    vq_cnorm   <<<KCB, 128, 0, stream>>>(cbw, cnormp);
    vq_bswz    <<<(KCB * DIM / 2) / 256, 256, 0, stream>>>(cbw, cbfr);
    vq_argmin  <<<NVEC / 128, 256, 0, stream>>>(x, (const unsigned short*)cbfr,
                                                cnormp, idxp);
    vq_scatter <<<BATCH * DIM, 256, 0, stream>>>(x, idxp, csump, csizep);
    vq_stats   <<<1, 1024, 0, stream>>>(ema_size, csizep, nes_out, scalp, perp_out);
    vq_codebook<<<(KCB * DIM) / 256, 256, 0, stream>>>(ema_sum, csump, nes_out,
                                                       scalp, ncb_out, nsum_out);
    vq_gather  <<<BATCH * DIM, 256, 0, stream>>>(x, ncb_out, idxp, q_out, scalp + 1);
    vq_final   <<<(NVEC + 255) / 256, 256, 0, stream>>>(idxp, scalp, ind_out, loss_out);
}
